// SimpleVoxelNet_43679817400609
// MI455X (gfx1250) — matrix operand order repaired and checked by reading, not compiled, not run on hardware
//
#include <hip/hip_runtime.h>

// ---------------- types ----------------
typedef __attribute__((ext_vector_type(16))) __bf16 v16bf;
typedef __attribute__((ext_vector_type(8)))  __bf16 v8bf;
typedef __attribute__((ext_vector_type(8)))  float  v8f;

// ---------------- problem constants ----------------
#define NXv   512
#define NYv   512
#define NZv   40
#define HWp   (512*512)           // BEV pixels
#define CAP   (1<<19)             // hash capacity (> 200k points)
#define EPSf  1e-5f

// =====================================================================
// Weight transform: OIHW fp32 -> [co][tap*Cin+ci] bf16
// =====================================================================
__global__ void k_wt(const float* __restrict__ K, __bf16* __restrict__ Wt, int Cin) {
    int i = blockIdx.x * blockDim.x + threadIdx.x;
    int tot = 128 * 9 * Cin;
    if (i >= tot) return;
    int k9  = 9 * Cin;
    int co  = i / k9;
    int rem = i - co * k9;
    int tap = rem / Cin;
    int ci  = rem - tap * Cin;
    int ky = tap / 3, kx = tap - ky * 3;
    Wt[i] = (__bf16)K[((co * Cin + ci) * 3 + ky) * 3 + kx];
}

// =====================================================================
// Points -> hash-table voxel accumulation
// =====================================================================
__global__ void k_points(const float* __restrict__ pts, int N,
                         int* __restrict__ keys, float* __restrict__ cnts,
                         float* __restrict__ sums, int* __restrict__ maxvox) {
    int i = blockIdx.x * blockDim.x + threadIdx.x;
    if (i >= N) return;
    float x = pts[i*4+0], y = pts[i*4+1], z = pts[i*4+2], w = pts[i*4+3];
    bool m = (x >= -51.2f) && (x < 51.2f) && (y >= -51.2f) && (y < 51.2f) &&
             (z >= -5.0f)  && (z < 3.0f);
    if (!m) return;
    int vx = (int)floorf((x + 51.2f) * 5.0f); vx = min(max(vx, 0), NXv - 1);
    int vy = (int)floorf((y + 51.2f) * 5.0f); vy = min(max(vy, 0), NYv - 1);
    int vz = (int)floorf((z +  5.0f) * 5.0f); vz = min(max(vz, 0), NZv - 1);
    int vox = vx + vy * NXv + vz * NXv * NYv;
    atomicMax(maxvox, vox);
    unsigned h = ((unsigned)vox * 2654435761u) & (CAP - 1);
    int slot = -1;
    for (int p = 0; p < CAP; ++p) {
        int k = atomicCAS(&keys[h], -1, vox);
        if (k == -1 || k == vox) { slot = (int)h; break; }
        h = (h + 1) & (CAP - 1);
    }
    if (slot < 0) return;
    atomicAdd(&cnts[slot], 1.0f);
    atomicAdd(&sums[slot*4+0], x);
    atomicAdd(&sums[slot*4+1], y);
    atomicAdd(&sums[slot*4+2], z);
    atomicAdd(&sums[slot*4+3], w);
}

// =====================================================================
// MLP layer-1 stats (h1 = mean@W1 + b1), masked over valid voxels
// =====================================================================
__global__ void k_mlp1(const int* __restrict__ keys, const float* __restrict__ cnts,
                       const float* __restrict__ sums, const int* __restrict__ maxvox,
                       const float* __restrict__ W1, const float* __restrict__ b1,
                       float* __restrict__ s1, float* __restrict__ ss1,
                       float* __restrict__ Mcnt) {
    __shared__ float ls[32], lss[32], lm;
    int t = threadIdx.x;
    if (t < 32) { ls[t] = 0.f; lss[t] = 0.f; }
    if (t == 0) lm = 0.f;
    __syncthreads();
    int slot = blockIdx.x * blockDim.x + t;
    int key  = (slot < CAP) ? keys[slot] : -1;
    bool valid = (key >= 0) && (key != *maxvox);
    if (valid) {
        float c = fmaxf(cnts[slot], 1.0f);
        float m0 = sums[slot*4+0]/c, m1 = sums[slot*4+1]/c;
        float m2 = sums[slot*4+2]/c, m3 = sums[slot*4+3]/c;
        atomicAdd(&lm, 1.0f);
        for (int j = 0; j < 32; ++j) {
            float h = m0*W1[j] + m1*W1[32+j] + m2*W1[64+j] + m3*W1[96+j] + b1[j];
            atomicAdd(&ls[j], h);
            atomicAdd(&lss[j], h*h);
        }
    }
    __syncthreads();
    if (t < 32) { atomicAdd(&s1[t], ls[t]); atomicAdd(&ss1[t], lss[t]); }
    if (t == 0) atomicAdd(Mcnt, lm);
}

// =====================================================================
// BN finalize: mu = s/cnt, rs = rsqrt(E[h^2]-mu^2 + eps)
// =====================================================================
__global__ void k_bnfin(const float* __restrict__ s, const float* __restrict__ ss,
                        const float* __restrict__ cntptr, float defcnt,
                        float* __restrict__ mu, float* __restrict__ rs, int n) {
    int t = blockIdx.x * blockDim.x + threadIdx.x;
    if (t >= n) return;
    float c = cntptr ? fmaxf(*cntptr, 1.0f) : defcnt;
    float m = s[t] / c;
    float v = ss[t] / c - m * m;
    mu[t] = m;
    rs[t] = rsqrtf(fmaxf(v, 0.0f) + EPSf);
}

// =====================================================================
// MLP layer-2 stats (a1 = relu(bn1(h1)); h2 = a1@W2 + b2)
// =====================================================================
__global__ void k_mlp2(const int* __restrict__ keys, const float* __restrict__ cnts,
                       const float* __restrict__ sums, const int* __restrict__ maxvox,
                       const float* __restrict__ W1, const float* __restrict__ b1,
                       const float* __restrict__ mu1, const float* __restrict__ rs1,
                       const float* __restrict__ g1, const float* __restrict__ be1,
                       const float* __restrict__ W2, const float* __restrict__ b2,
                       float* __restrict__ s2, float* __restrict__ ss2) {
    __shared__ float ls[64], lss[64];
    int t = threadIdx.x;
    if (t < 64) { ls[t] = 0.f; lss[t] = 0.f; }
    __syncthreads();
    int slot = blockIdx.x * blockDim.x + t;
    int key  = (slot < CAP) ? keys[slot] : -1;
    bool valid = (key >= 0) && (key != *maxvox);
    if (valid) {
        float c = fmaxf(cnts[slot], 1.0f);
        float m0 = sums[slot*4+0]/c, m1 = sums[slot*4+1]/c;
        float m2 = sums[slot*4+2]/c, m3 = sums[slot*4+3]/c;
        float a1[32];
        for (int j = 0; j < 32; ++j) {
            float h = m0*W1[j] + m1*W1[32+j] + m2*W1[64+j] + m3*W1[96+j] + b1[j];
            a1[j] = fmaxf((h - mu1[j]) * rs1[j] * g1[j] + be1[j], 0.0f);
        }
        for (int j = 0; j < 64; ++j) {
            float h = b2[j];
            for (int i = 0; i < 32; ++i) h += a1[i] * W2[i*64 + j];
            atomicAdd(&ls[j], h);
            atomicAdd(&lss[j], h*h);
        }
    }
    __syncthreads();
    if (t < 64) { atomicAdd(&s2[t], ls[t]); atomicAdd(&ss2[t], lss[t]); }
}

// =====================================================================
// MLP final + BEV scatter (64 channels, fp32 atomics)
// =====================================================================
__global__ void k_mlp3(const int* __restrict__ keys, const float* __restrict__ cnts,
                       const float* __restrict__ sums, const int* __restrict__ maxvox,
                       const float* __restrict__ W1, const float* __restrict__ b1,
                       const float* __restrict__ mu1, const float* __restrict__ rs1,
                       const float* __restrict__ g1, const float* __restrict__ be1,
                       const float* __restrict__ W2, const float* __restrict__ b2,
                       const float* __restrict__ mu2, const float* __restrict__ rs2,
                       const float* __restrict__ g2, const float* __restrict__ be2,
                       float* __restrict__ bev) {
    int slot = blockIdx.x * blockDim.x + threadIdx.x;
    if (slot >= CAP) return;
    int key = keys[slot];
    if (key < 0 || key == *maxvox) return;
    float c = fmaxf(cnts[slot], 1.0f);
    float m0 = sums[slot*4+0]/c, m1 = sums[slot*4+1]/c;
    float m2 = sums[slot*4+2]/c, m3 = sums[slot*4+3]/c;
    float a1[32];
    for (int j = 0; j < 32; ++j) {
        float h = m0*W1[j] + m1*W1[32+j] + m2*W1[64+j] + m3*W1[96+j] + b1[j];
        a1[j] = fmaxf((h - mu1[j]) * rs1[j] * g1[j] + be1[j], 0.0f);
    }
    int pix = key % (NXv * NYv);     // yc*NX + xc
    for (int j = 0; j < 64; ++j) {
        float h = b2[j];
        for (int i = 0; i < 32; ++i) h += a1[i] * W2[i*64 + j];
        float a2 = fmaxf((h - mu2[j]) * rs2[j] * g2[j] + be2[j], 0.0f);
        atomicAdd(&bev[(size_t)pix * 64 + j], a2);
    }
}

// =====================================================================
// fp32 -> bf16 convert (HWC)
// =====================================================================
__global__ void k_tobf16(const float* __restrict__ s, __bf16* __restrict__ d, int n) {
    int i = blockIdx.x * blockDim.x + threadIdx.x;
    if (i < n) d[i] = (__bf16)s[i];
}

// =====================================================================
// 3x3 conv, implicit GEMM via v_wmma_f32_16x16x32_bf16.
// Input HWC bf16, weights [co][tap*Cin+ci] bf16, output HWC fp32 (Cout=128).
// Block = 256 thr = 8 waves, covers 64 pixels x 128 couts of one row.
// Input strip (3 rows x 66 px x Cin, halo pre-zeroed) staged in LDS once;
// each wave holds 4 pixel-group accumulators so each B fragment feeds
// 4 WMMAs and A comes from LDS (ds_load_b128).
// Also accumulates per-channel sum / sumsq for BN (LDS then global atomics).
// =====================================================================
__global__ __launch_bounds__(256)
void k_conv(const __bf16* __restrict__ in, int Cin,
            const __bf16* __restrict__ Wt, const float* __restrict__ bias,
            float* __restrict__ out,
            float* __restrict__ csum, float* __restrict__ css) {
    __shared__ __align__(16) __bf16 stage[3 * 66 * 128];  // max Cin=128: 50.7KB
    __shared__ float ssum[128], ssq[128];
    int t = threadIdx.x;
    if (t < 128) { ssum[t] = 0.f; ssq[t] = 0.f; }

    int pix0 = blockIdx.x * 64;
    int y  = pix0 >> 9;
    int x0 = pix0 & 511;

    // ---- cooperative stage of input strip [3 rows][66 px][Cin], halo zeroed ----
    int cin8   = Cin >> 3;                 // v8bf chunks per pixel
    int perRow = 66 * cin8;
    int total  = 3 * perRow;
    for (int c = t; c < total; c += 256) {
        int r   = c / perRow;
        int rem = c - r * perRow;
        int px  = rem / cin8;
        int co  = (rem - px * cin8) * 8;
        int yy  = y + r - 1;
        int xx  = x0 + px - 1;
        v8bf val;
        if (yy >= 0 && yy < 512 && xx >= 0 && xx < 512) {
            val = *(const v8bf*)(in + (size_t)(yy * 512 + xx) * Cin + co);
        } else {
#pragma unroll
            for (int e = 0; e < 8; ++e) val[e] = (__bf16)0.0f;
        }
        *(v8bf*)(stage + (size_t)(r * 66 + px) * Cin + co) = val;
    }
    __syncthreads();

    int wave = t >> 5, lane = t & 31;
    int m = lane & 15, sel = lane >> 4;
    int cb = wave * 16;
    int K9 = 9 * Cin;
    const __bf16* wbase = Wt + (size_t)(cb + m) * K9;

    v8f acc0, acc1, acc2, acc3;
#pragma unroll
    for (int r = 0; r < 8; ++r) { acc0[r] = 0.f; acc1[r] = 0.f; acc2[r] = 0.f; acc3[r] = 0.f; }

    for (int tap = 0; tap < 9; ++tap) {
        int r  = tap / 3;                   // stage row (dy+1)
        int dx = tap - r * 3;               // 0..2 == (kx), stage px offset
        const __bf16* brow = wbase + tap * Cin;
        if (tap < 8) __builtin_prefetch(brow + Cin, 0, 1);   // global_prefetch_b8
        for (int c0 = 0; c0 < Cin; c0 += 32) {
            // B fragment: 32x16 bf16; lane half selects K {0-7,16-23} vs {8-15,24-31} of column n
            const v8bf* bp = (const v8bf*)(brow + c0 + sel * 8);
            v8bf b0 = bp[0], b1 = bp[2];
            v16bf b;
#pragma unroll
            for (int e = 0; e < 8; ++e) { b[e] = b0[e]; b[e + 8] = b1[e]; }
            // 4 pixel groups reuse this B fragment
#pragma unroll
            for (int g = 0; g < 4; ++g) {
                const __bf16* ab = stage + (size_t)(r * 66 + g * 16 + m + dx) * Cin + c0;
                v8bf lo = *(const v8bf*)(ab + sel * 8);
                v8bf hi = *(const v8bf*)(ab + 16 + sel * 8);
                v16bf a;
#pragma unroll
                for (int e = 0; e < 8; ++e) { a[e] = lo[e]; a[e + 8] = hi[e]; }
                if (g == 0)      acc0 = __builtin_amdgcn_wmma_f32_16x16x32_bf16(false, a, false, b, (short)0, acc0, false, false);
                else if (g == 1) acc1 = __builtin_amdgcn_wmma_f32_16x16x32_bf16(false, a, false, b, (short)0, acc1, false, false);
                else if (g == 2) acc2 = __builtin_amdgcn_wmma_f32_16x16x32_bf16(false, a, false, b, (short)0, acc2, false, false);
                else             acc3 = __builtin_amdgcn_wmma_f32_16x16x32_bf16(false, a, false, b, (short)0, acc3, false, false);
            }
        }
    }

    float bbias = bias[cb + m];
    float ps = 0.f, pq = 0.f;
#pragma unroll
    for (int g = 0; g < 4; ++g) {
        v8f acc = (g == 0) ? acc0 : (g == 1) ? acc1 : (g == 2) ? acc2 : acc3;
#pragma unroll
        for (int rr = 0; rr < 8; ++rr) {
            float v = acc[rr] + bbias;      // lane row M = sel*8 + rr
            out[(size_t)(y * 512 + x0 + g * 16 + sel * 8 + rr) * 128 + cb + m] = v;
            ps += v; pq += v * v;
        }
    }
    atomicAdd(&ssum[cb + m], ps);
    atomicAdd(&ssq[cb + m], pq);
    __syncthreads();
    if (t < 128) { atomicAdd(&csum[t], ssum[t]); atomicAdd(&css[t], ssq[t]); }
}

// =====================================================================
// BN + ReLU; write bf16 HWC for next layer, or fp32 NCHW into d_out.
// =====================================================================
__global__ void k_bnrelu(const float* __restrict__ raw,
                         const float* __restrict__ mu, const float* __restrict__ rs,
                         const float* __restrict__ g,  const float* __restrict__ be,
                         __bf16* __restrict__ nb, float* __restrict__ fout, int last) {
    int i = blockIdx.x * blockDim.x + threadIdx.x;   // over HW*128
    int c = i & 127;
    float v = (raw[i] - mu[c]) * rs[c] * g[c] + be[c];
    v = fmaxf(v, 0.0f);
    if (last) fout[(size_t)c * HWp + (i >> 7)] = v;
    else      nb[i] = (__bf16)v;
}

// =====================================================================
// Launcher
// =====================================================================
extern "C" void kernel_launch(void* const* d_in, const int* in_sizes, int n_in,
                              void* d_out, int out_size, void* d_ws, size_t ws_size,
                              hipStream_t stream) {
    const float* pts = (const float*)d_in[0];
    int N = in_sizes[0] / 4;
    const float* W1  = (const float*)d_in[1];
    const float* b1  = (const float*)d_in[2];
    const float* g1  = (const float*)d_in[3];
    const float* be1 = (const float*)d_in[4];
    const float* W2  = (const float*)d_in[5];
    const float* b2  = (const float*)d_in[6];
    const float* g2  = (const float*)d_in[7];
    const float* be2 = (const float*)d_in[8];
    const float* K1  = (const float*)d_in[9];
    const float* cb1 = (const float*)d_in[10];
    const float* cg1 = (const float*)d_in[11];
    const float* cbe1= (const float*)d_in[12];
    const float* K2  = (const float*)d_in[13];
    const float* cb2 = (const float*)d_in[14];
    const float* cg2 = (const float*)d_in[15];
    const float* cbe2= (const float*)d_in[16];
    const float* K3  = (const float*)d_in[17];
    const float* cb3 = (const float*)d_in[18];
    const float* cg3 = (const float*)d_in[19];
    const float* cbe3= (const float*)d_in[20];
    float* fout = (float*)d_out;

    // ---- workspace carve-up ----
    char* w = (char*)d_ws;
    size_t off = 0;
    int*    keys = (int*)  (w + off); off += (size_t)CAP * 4;
    float*  cnts = (float*)(w + off); off += (size_t)CAP * 4;
    float*  sums = (float*)(w + off); off += (size_t)CAP * 16;
    float*  scal = (float*)(w + off); off += 16384;
    __bf16* wt1  = (__bf16*)(w + off); off += (size_t)128 * 576 * 2;
    __bf16* wt2  = (__bf16*)(w + off); off += (size_t)128 * 1152 * 2;
    __bf16* wt3  = (__bf16*)(w + off); off += (size_t)128 * 1152 * 2;
    off = (off + 255) & ~(size_t)255;
    float*  bev  = (float*)(w + off); off += (size_t)HWp * 64 * 4;
    float*  raw  = (float*)(w + off); off += (size_t)HWp * 128 * 4;
    __bf16* actA = (__bf16*)(w + off); off += (size_t)HWp * 128 * 2;
    __bf16* actB = (__bf16*)(w + off); off += (size_t)HWp * 128 * 2;

    int*   maxvox = (int*)scal;
    float* Mcnt   = scal + 1;
    float* s1  = scal + 8;    float* ss1 = scal + 40;
    float* mu1 = scal + 72;   float* rs1 = scal + 104;
    float* s2  = scal + 136;  float* ss2 = scal + 200;
    float* mu2 = scal + 264;  float* rs2 = scal + 328;
    float* cst = scal + 392;  // per conv layer l: [cs,css,cmu,crs] x 128
#define CS(l)  (cst + (l)*512)
#define CSS(l) (cst + (l)*512 + 128)
#define CMU(l) (cst + (l)*512 + 256)
#define CRS(l) (cst + (l)*512 + 384)

    // ---- per-call initialization (graph-capture safe) ----
    hipMemsetAsync(keys, 0xFF, (size_t)CAP * 4, stream);
    hipMemsetAsync(cnts, 0,    (size_t)CAP * 4 + (size_t)CAP * 16, stream);
    hipMemsetAsync(scal, 0,    16384, stream);
    hipMemsetAsync(maxvox, 0xFF, 4, stream);
    hipMemsetAsync(bev,  0,    (size_t)HWp * 64 * 4, stream);

    // ---- weight transforms ----
    k_wt<<<(128*9*64  + 255)/256, 256, 0, stream>>>(K1, wt1, 64);
    k_wt<<<(128*9*128 + 255)/256, 256, 0, stream>>>(K2, wt2, 128);
    k_wt<<<(128*9*128 + 255)/256, 256, 0, stream>>>(K3, wt3, 128);

    // ---- voxelize + MLP + scatter ----
    k_points<<<(N + 255)/256, 256, 0, stream>>>(pts, N, keys, cnts, sums, maxvox);
    k_mlp1<<<CAP/256, 256, 0, stream>>>(keys, cnts, sums, maxvox, W1, b1, s1, ss1, Mcnt);
    k_bnfin<<<1, 32, 0, stream>>>(s1, ss1, Mcnt, 1.0f, mu1, rs1, 32);
    k_mlp2<<<CAP/256, 256, 0, stream>>>(keys, cnts, sums, maxvox,
                                        W1, b1, mu1, rs1, g1, be1, W2, b2, s2, ss2);
    k_bnfin<<<1, 64, 0, stream>>>(s2, ss2, Mcnt, 1.0f, mu2, rs2, 64);
    k_mlp3<<<CAP/256, 256, 0, stream>>>(keys, cnts, sums, maxvox,
                                        W1, b1, mu1, rs1, g1, be1,
                                        W2, b2, mu2, rs2, g2, be2, bev);
    k_tobf16<<<(HWp*64)/256, 256, 0, stream>>>(bev, actA, HWp * 64);

    const float spatial = (float)HWp;   // conv BN count = H*W (batch 1)

    // ---- conv1 (64 -> 128) ----
    k_conv<<<HWp/64, 256, 0, stream>>>(actA, 64, wt1, cb1, raw, CS(0), CSS(0));
    k_bnfin<<<1, 128, 0, stream>>>(CS(0), CSS(0), nullptr, spatial, CMU(0), CRS(0), 128);
    k_bnrelu<<<(HWp*128)/256, 256, 0, stream>>>(raw, CMU(0), CRS(0), cg1, cbe1, actB, nullptr, 0);

    // ---- conv2 (128 -> 128) ----
    k_conv<<<HWp/64, 256, 0, stream>>>(actB, 128, wt2, cb2, raw, CS(1), CSS(1));
    k_bnfin<<<1, 128, 0, stream>>>(CS(1), CSS(1), nullptr, spatial, CMU(1), CRS(1), 128);
    k_bnrelu<<<(HWp*128)/256, 256, 0, stream>>>(raw, CMU(1), CRS(1), cg2, cbe2, actA, nullptr, 0);

    // ---- conv3 (128 -> 128), BN+ReLU straight into d_out (NCHW fp32) ----
    k_conv<<<HWp/64, 256, 0, stream>>>(actA, 128, wt3, cb3, raw, CS(2), CSS(2));
    k_bnfin<<<1, 128, 0, stream>>>(CS(2), CSS(2), nullptr, spatial, CMU(2), CRS(2), 128);
    k_bnrelu<<<(HWp*128)/256, 256, 0, stream>>>(raw, CMU(2), CRS(2), cg3, cbe3, nullptr, fout, 1);
}
